// OmegaRelGraphConv_35450660061270
// MI455X (gfx1250) — matrix operand order repaired and checked by reading, not compiled, not run on hardware
//
#include <hip/hip_runtime.h>

#define D 128
#define RRELU_SLOPE 0.22916666666666666f  // 11/48

typedef __attribute__((ext_vector_type(16))) __bf16    v16bf;
typedef __attribute__((ext_vector_type(8)))  float     v8f;
typedef __attribute__((ext_vector_type(8)))  unsigned  v8u;
typedef __attribute__((ext_vector_type(4)))  float     f32x4;
typedef __attribute__((ext_vector_type(4)))  unsigned  u32x4;

// Convert two f32 to packed 2xbf16 (round-to-nearest, ties toward zero) in 3 VALU ops:
// two v_add_u32 + one v_perm_b32 picking the high halves.
__device__ inline unsigned pack2_bf16(float f0, float f1) {
  unsigned u0 = __builtin_bit_cast(unsigned, f0) + 0x7FFFu;
  unsigned u1 = __builtin_bit_cast(unsigned, f1) + 0x7FFFu;
  // result byte0,1 = u0 bytes 2,3 (low bf16); byte2,3 = u1 bytes 2,3 (high bf16)
  return __builtin_amdgcn_perm(u1, u0, 0x07060302u);
}

__device__ inline v8f wmma_bf16(v16bf a, v16bf b, v8f c) {
  return __builtin_amdgcn_wmma_f32_16x16x32_bf16(
      /*neg_a=*/false, a, /*neg_b=*/false, b,
      /*c_mod=*/(short)0, c, /*reuse_a=*/false, /*reuse_b=*/false);
}

// Pack 16 f32 (two K-octets: x[0..3],y[0..3] = K k0..k0+7 ; z,w = K k1..k1+7)
// into the wave32 16-bit A-operand register layout.
__device__ inline v16bf make_a(f32x4 x, f32x4 y, f32x4 z, f32x4 w) {
  v8u au;
  au[0] = pack2_bf16(x[0], x[1]);
  au[1] = pack2_bf16(x[2], x[3]);
  au[2] = pack2_bf16(y[0], y[1]);
  au[3] = pack2_bf16(y[2], y[3]);
  au[4] = pack2_bf16(z[0], z[1]);
  au[5] = pack2_bf16(z[2], z[3]);
  au[6] = pack2_bf16(w[0], w[1]);
  au[7] = pack2_bf16(w[2], w[3]);
  return __builtin_bit_cast(v16bf, au);
}

// ---------------- utility kernels ----------------

__global__ void zero_f32(float* __restrict__ p, int n) {
  int i = blockIdx.x * blockDim.x + threadIdx.x;
  int stride = gridDim.x * blockDim.x;
  for (; i < n; i += stride) p[i] = 0.0f;
}

__global__ void deg_count(const int* __restrict__ dst, float* __restrict__ deg, int E) {
  int i = blockIdx.x * blockDim.x + threadIdx.x;
  if (i < E) atomicAdd(&deg[dst[i]], 1.0f);
}

__global__ void cvt_bf16(const float* __restrict__ w, unsigned* __restrict__ wb, int npairs) {
  int i = blockIdx.x * blockDim.x + threadIdx.x;
  if (i < npairs) wb[i] = pack2_bf16(w[2 * i], w[2 * i + 1]);
}

// ---------------- edge message GEMM + scatter ----------------
// One wave handles a 16-edge tile: A = (h[src]+rel)[16x128] in bf16,
// B = w1 (bf16, staged in LDS), C/D = 16x128 f32 -> atomicAdd into agg[dst].
__global__ void edge_msg(const float* __restrict__ h,
                         const float* __restrict__ rel,
                         const int*   __restrict__ src,
                         const int*   __restrict__ dst,
                         const __bf16* __restrict__ w1b,
                         float* __restrict__ agg,
                         int E) {
  __shared__ __attribute__((aligned(32))) __bf16 w1s[D * D];   // 32 KB of 320 KB WGP LDS
  for (int i = threadIdx.x; i < (D * D) / 8; i += blockDim.x)
    ((u32x4*)w1s)[i] = ((const u32x4*)w1b)[i];
  __syncthreads();

  const int lane = threadIdx.x & 31;
  const int wave = threadIdx.x >> 5;
  const int tile = blockIdx.x * (blockDim.x >> 5) + wave;
  const int e0   = tile * 16;
  if (e0 >= E) return;                         // wave-uniform

  const int hi = lane >> 4;                    // 0: K base, 1: K+8 half
  const int m  = lane & 15;                    // A row / D column owned by lane
  int e  = e0 + m;
  int ec = (e < E) ? e : e0;                   // clamp tail (results discarded)
  const int s = src[ec];
  const f32x4* h4 = (const f32x4*)(h + s * D);         // row, float4 units
  const f32x4* r4 = (const f32x4*)(rel + (size_t)ec * D);

  v8f acc[8];
  const v8f vzero = {0.f,0.f,0.f,0.f,0.f,0.f,0.f,0.f};
#pragma unroll
  for (int t = 0; t < 8; ++t) acc[t] = vzero;

#pragma unroll
  for (int kt = 0; kt < 4; ++kt) {
    const int q0 = kt * 8 + hi * 2;            // float4 index of K=k0
    const int q1 = q0 + 4;                     // float4 index of K=k0+16
    f32x4 x = h4[q0]     + r4[q0];             // v_pk_add_f32
    f32x4 y = h4[q0 + 1] + r4[q0 + 1];
    f32x4 z = h4[q1]     + r4[q1];
    f32x4 w = h4[q1 + 1] + r4[q1 + 1];
    v16bf a = make_a(x, y, z, w);

    const __bf16* bs = w1s + (kt * 32 + hi * 8) * D + m;  // B col m, K = kt*32+8*hi+{0..7,16..23}
#pragma unroll
    for (int nt = 0; nt < 8; ++nt) {
      v16bf b;
#pragma unroll
      for (int s2 = 0; s2 < 8; ++s2) {
        b[s2]     = bs[s2 * D + nt * 16];
        b[8 + s2] = bs[(16 + s2) * D + nt * 16];
      }
      acc[nt] = wmma_bf16(a, b, acc[nt]);
    }
  }

  // scatter: acc[nt][g] is row (g + 8*hi), col (nt*16 + m)
  int rowoff[8];
#pragma unroll
  for (int g = 0; g < 8; ++g) {
    int eg = e0 + g + 8 * hi;
    rowoff[g] = (eg < E) ? dst[eg] * D : -1;
  }
#pragma unroll
  for (int nt = 0; nt < 8; ++nt) {
    const int col = nt * 16 + m;
#pragma unroll
    for (int g = 0; g < 8; ++g) {
      if (rowoff[g] >= 0)
        atomicAdd(&agg[rowoff[g] + col], acc[nt][g]);  // L2-resident atomics
    }
  }
}

// ---------------- node epilogue: self-loop GEMMs + mean + rrelu ----------------
__global__ void node_update(const float* __restrict__ h,
                            const float* __restrict__ agg,
                            const float* __restrict__ deg,
                            const __bf16* __restrict__ w2b,
                            const __bf16* __restrict__ w3b,
                            float* __restrict__ out,
                            int N) {
  const int lane = threadIdx.x & 31;
  const int wave = threadIdx.x >> 5;
  const int tile = blockIdx.x * (blockDim.x >> 5) + wave;
  const int n0   = tile * 16;
  if (n0 >= N) return;                         // wave-uniform

  const int hi = lane >> 4;
  const int m  = lane & 15;
  int node  = n0 + m;
  int nodec = (node < N) ? node : n0;
  const f32x4* h4 = (const f32x4*)(h + nodec * D);

  // A for all 4 K-steps kept in registers (32 VGPRs)
  v16bf a[4];
#pragma unroll
  for (int kt = 0; kt < 4; ++kt) {
    const int q0 = kt * 8 + hi * 2;
    const int q1 = q0 + 4;
    a[kt] = make_a(h4[q0], h4[q0 + 1], h4[q1], h4[q1 + 1]);
  }

  // rows this lane owns in the C/D layout
  int   rowoff[8];
  float rowdeg[8];
#pragma unroll
  for (int g = 0; g < 8; ++g) {
    int ng = n0 + g + 8 * hi;
    rowoff[g] = (ng < N) ? ng * D : -1;
    rowdeg[g] = (ng < N) ? deg[ng] : 1.0f;
  }

  const v8f vzero = {0.f,0.f,0.f,0.f,0.f,0.f,0.f,0.f};
#pragma unroll
  for (int nt = 0; nt < 8; ++nt) {
    v8f acc2 = vzero, acc3 = vzero;
#pragma unroll
    for (int kt = 0; kt < 4; ++kt) {
      const int boff = (kt * 32 + hi * 8) * D + nt * 16 + m;
      v16bf b2, b3;  // L2-hot
#pragma unroll
      for (int s2 = 0; s2 < 8; ++s2) {
        b2[s2]     = w2b[boff + s2 * D];
        b2[8 + s2] = w2b[boff + (16 + s2) * D];
        b3[s2]     = w3b[boff + s2 * D];
        b3[8 + s2] = w3b[boff + (16 + s2) * D];
      }
      acc2 = wmma_bf16(a[kt], b2, acc2);
      acc3 = wmma_bf16(a[kt], b3, acc3);
    }
    const int col = nt * 16 + m;
#pragma unroll
    for (int g = 0; g < 8; ++g) {
      if (rowoff[g] >= 0) {
        float dg   = rowdeg[g];
        float self = (dg == 0.0f) ? acc3[g] : acc2[g];
        float av   = agg[rowoff[g] + col] / fmaxf(dg, 1.0f);
        float x    = av + self;
        out[rowoff[g] + col] = (x > 0.0f) ? x : RRELU_SLOPE * x;
      }
    }
  }
}

// ---------------- host ----------------

extern "C" void kernel_launch(void* const* d_in, const int* in_sizes, int n_in,
                              void* d_out, int out_size, void* d_ws, size_t ws_size,
                              hipStream_t stream) {
  const float* ent  = (const float*)d_in[0];
  const float* rel  = (const float*)d_in[1];
  const int*   src  = (const int*)  d_in[2];
  const int*   dst  = (const int*)  d_in[3];
  const float* w1_0 = (const float*)d_in[4];
  const float* w2_0 = (const float*)d_in[5];
  const float* w3_0 = (const float*)d_in[6];
  const float* w1_1 = (const float*)d_in[7];
  const float* w2_1 = (const float*)d_in[8];
  const float* w3_1 = (const float*)d_in[9];
  float* out = (float*)d_out;

  const int N = in_sizes[0] / D;
  const int E = in_sizes[2];

  char* ws = (char*)d_ws;
  size_t off = 0;
  auto wsalloc = [&](size_t bytes) -> void* {
    void* p = ws + off;
    off += (bytes + 255) & ~(size_t)255;
    return p;
  };
  float*  deg = (float*)wsalloc((size_t)N * sizeof(float));
  float*  agg = (float*)wsalloc((size_t)N * D * sizeof(float));
  float*  h1  = (float*)wsalloc((size_t)N * D * sizeof(float));
  __bf16* w1_0b = (__bf16*)wsalloc(D * D * sizeof(__bf16));
  __bf16* w2_0b = (__bf16*)wsalloc(D * D * sizeof(__bf16));
  __bf16* w3_0b = (__bf16*)wsalloc(D * D * sizeof(__bf16));
  __bf16* w1_1b = (__bf16*)wsalloc(D * D * sizeof(__bf16));
  __bf16* w2_1b = (__bf16*)wsalloc(D * D * sizeof(__bf16));
  __bf16* w3_1b = (__bf16*)wsalloc(D * D * sizeof(__bf16));

  const int TPB = 256;                 // 8 wave32 waves
  const int etiles  = (E + 15) / 16;
  const int ntiles  = (N + 15) / 16;
  const int eblocks = (etiles + 7) / 8;
  const int nblocks = (ntiles + 7) / 8;
  const int wpairs  = D * D / 2;

  // init
  zero_f32<<<256, TPB, 0, stream>>>(deg, N);
  zero_f32<<<2048, TPB, 0, stream>>>(agg, N * D);
  deg_count<<<(E + TPB - 1) / TPB, TPB, 0, stream>>>(dst, deg, E);
  cvt_bf16<<<(wpairs + TPB - 1) / TPB, TPB, 0, stream>>>(w1_0, (unsigned*)w1_0b, wpairs);
  cvt_bf16<<<(wpairs + TPB - 1) / TPB, TPB, 0, stream>>>(w2_0, (unsigned*)w2_0b, wpairs);
  cvt_bf16<<<(wpairs + TPB - 1) / TPB, TPB, 0, stream>>>(w3_0, (unsigned*)w3_0b, wpairs);
  cvt_bf16<<<(wpairs + TPB - 1) / TPB, TPB, 0, stream>>>(w1_1, (unsigned*)w1_1b, wpairs);
  cvt_bf16<<<(wpairs + TPB - 1) / TPB, TPB, 0, stream>>>(w2_1, (unsigned*)w2_1b, wpairs);
  cvt_bf16<<<(wpairs + TPB - 1) / TPB, TPB, 0, stream>>>(w3_1, (unsigned*)w3_1b, wpairs);

  // layer 1
  edge_msg<<<eblocks, TPB, 0, stream>>>(ent, rel, src, dst, w1_0b, agg, E);
  node_update<<<nblocks, TPB, 0, stream>>>(ent, agg, deg, w2_0b, w3_0b, h1, N);

  // layer 2
  zero_f32<<<2048, TPB, 0, stream>>>(agg, N * D);
  edge_msg<<<eblocks, TPB, 0, stream>>>(h1, rel, src, dst, w1_1b, agg, E);
  node_update<<<nblocks, TPB, 0, stream>>>(h1, agg, deg, w2_1b, w3_1b, out, N);
}
